// VectorQuantizer_87540023427099
// MI455X (gfx1250) — compile-verified, matrix operand order checked by reading, not run on hardware
//
#include <hip/hip_runtime.h>
#include <hip/hip_bf16.h>

typedef __attribute__((ext_vector_type(16))) _Float16 v16h;
typedef __attribute__((ext_vector_type(8)))  _Float16 v8h;
typedef __attribute__((ext_vector_type(8)))  float    v8f;

#define N_E    8192
#define EDIM   256
#define NTOT   32768          // 32 * 32 * 32 vectors
#define TILE_N 64
#define TILE_M 64
#define WSCALE 8192.0f        // codebook pre-scale so f16 entries are ~U(-1,1)

#if __has_builtin(__builtin_amdgcn_sched_barrier)
#define SCHED_FENCE() __builtin_amdgcn_sched_barrier(0)
#else
#define SCHED_FENCE()
#endif

// ---------------------------------------------------------------------------
// Prep: codebook -> f16 (scaled), plus fp32 squared norms of scaled rows.
// ---------------------------------------------------------------------------
__global__ void vq_prep_w(const float* __restrict__ W,
                          _Float16* __restrict__ wsf16,
                          float* __restrict__ wn2) {
  const int lane = threadIdx.x & 31;
  const int grp  = threadIdx.x >> 5;             // 0..7
  const int code = blockIdx.x * 8 + grp;         // 0..8191
  const float*    src = W     + (size_t)code * EDIM + lane * 8;
  _Float16*       dst = wsf16 + (size_t)code * EDIM + lane * 8;
  float s = 0.0f;
#pragma unroll
  for (int i = 0; i < 8; ++i) {
    float vs = src[i] * WSCALE;
    s += vs * vs;
    dst[i] = (_Float16)vs;
  }
#pragma unroll
  for (int off = 16; off >= 1; off >>= 1) s += __shfl_xor(s, off, 32);
  if (lane == 0) wn2[code] = s;
}

// ---------------------------------------------------------------------------
// Prep: z (B,C,H,W) fp32 -> zf16 (N=B*H*W, C) f16 row-major.
// ---------------------------------------------------------------------------
__global__ void vq_prep_z(const float* __restrict__ z,
                          _Float16* __restrict__ zf16) {
  const int b  = blockIdx.x >> 5;
  const int h  = blockIdx.x & 31;
  const int cl = threadIdx.x >> 5;               // 0..7
  const int w  = threadIdx.x & 31;
#pragma unroll 4
  for (int cc = 0; cc < 32; ++cc) {
    const int c = cc * 8 + cl;
    float v = z[(size_t)(b * EDIM + c) * 1024 + h * 32 + w];
    zf16[(size_t)((b * 32 + h) * 32 + w) * EDIM + c] = (_Float16)v;
  }
}

// ---------------------------------------------------------------------------
// Fused GEMM (z @ Ws^T via v_wmma_f32_16x16x32_f16) + running argmin.
// 512 blocks x 256 threads (8 waves). Block owns 64 z-rows; sweeps all 8192
// codes in 64-wide LDS-staged tiles. Wave = (n-strip 0..3, m-half 0..1).
// A sched_barrier splits each m-tile into [32 batched ds_load_b128] then
// [16 WMMAs], so the waitcnt inserter emits partial s_wait_dscnt waits and
// LDS latency overlaps the WMMA chains. waves_per_eu(2) raises the VGPR
// budget so all 32 fragments stay resident (no remat back to load-per-wmma).
// ---------------------------------------------------------------------------
__global__ __launch_bounds__(256)
__attribute__((amdgpu_waves_per_eu(2)))
void vq_argmin_wmma(
    const _Float16* __restrict__ zf16, const _Float16* __restrict__ wsf16,
    const float* __restrict__ wn2, int* __restrict__ idxout) {
  __shared__ _Float16 Bs[TILE_M * EDIM];         // 32 KB codebook tile
  __shared__ float    redv[4][16][2];
  __shared__ int      redi[4][16][2];

  const int tid   = threadIdx.x;
  const int lane  = tid & 31;
  const int wid   = tid >> 5;                    // 0..7
  const int strip = wid & 3;                     // which 16-row n strip
  const int mhalf = wid >> 2;                    // which 32-col m half
  const int l16   = lane & 15;
  const int sel8  = (lane >> 4) * 8;             // f16 A/B lane K-phase

  const int nbase = blockIdx.x * TILE_N;

  // Resident A fragments: this wave's 16-row strip, all 8 K-chunks (64 VGPRs).
  v16h aFrag[8];
  {
    const _Float16* arow = zf16 + (size_t)(nbase + strip * 16 + l16) * EDIM + sel8;
#pragma unroll
    for (int kc = 0; kc < 8; ++kc) {
      const _Float16* p = arow + kc * 32;
      v8h lo = *(const v8h*)p;
      v8h hi = *(const v8h*)(p + 16);
      aFrag[kc] = __builtin_shufflevector(lo, hi, 0, 1, 2, 3, 4, 5, 6, 7,
                                          8, 9, 10, 11, 12, 13, 14, 15);
    }
  }

  float minv[8];
  int   mini[8];
#pragma unroll
  for (int r = 0; r < 8; ++r) { minv[r] = __builtin_inff(); mini[r] = 0; }

  for (int mbase = 0; mbase < N_E; mbase += TILE_M) {
    __syncthreads();
    {  // cooperative 64x256 f16 tile load, 16B per op, fully coalesced
      const uint4* src = (const uint4*)(wsf16 + (size_t)mbase * EDIM);
      uint4*       dst = (uint4*)Bs;
#pragma unroll
      for (int i = 0; i < 8; ++i) dst[tid + i * 256] = src[tid + i * 256];
      if (mbase + TILE_M < N_E)  // hint next tile into cache (global_prefetch)
        __builtin_prefetch(wsf16 + (size_t)(mbase + TILE_M) * EDIM + tid * 16, 0, 0);
    }
    __syncthreads();

    // ---- Batch-load ALL 32 b128 B fragments (interleaved to match WMMA order)
    const _Float16* brow0 = Bs + (size_t)(mhalf * 32 + l16) * EDIM + sel8;
    const _Float16* brow1 = brow0 + (size_t)16 * EDIM;
    v16h b0[8], b1[8];
    SCHED_FENCE();
#pragma unroll
    for (int kc = 0; kc < 8; ++kc) {
      const _Float16* p0 = brow0 + kc * 32;
      v8h lo0 = *(const v8h*)p0;
      v8h hi0 = *(const v8h*)(p0 + 16);
      b0[kc] = __builtin_shufflevector(lo0, hi0, 0, 1, 2, 3, 4, 5, 6, 7,
                                       8, 9, 10, 11, 12, 13, 14, 15);
      const _Float16* p1 = brow1 + kc * 32;
      v8h lo1 = *(const v8h*)p1;
      v8h hi1 = *(const v8h*)(p1 + 16);
      b1[kc] = __builtin_shufflevector(lo1, hi1, 0, 1, 2, 3, 4, 5, 6, 7,
                                       8, 9, 10, 11, 12, 13, 14, 15);
    }
    SCHED_FENCE();   // <-- no WMMA may move above; no ds_load may sink below

    // ---- Two independent WMMA accumulation chains (interleaved)
    v8f acc0 = {}, acc1 = {};
#pragma unroll
    for (int kc = 0; kc < 8; ++kc) {
      acc0 = __builtin_amdgcn_wmma_f32_16x16x32_f16(
          false, aFrag[kc], false, b0[kc], (short)0, acc0, false, false);
      acc1 = __builtin_amdgcn_wmma_f32_16x16x32_f16(
          false, aFrag[kc], false, b1[kc], (short)0, acc1, false, false);
    }
    SCHED_FENCE();

    // ---- Epilogue: running argmin over this lane's two columns
    const int   m0 = mbase + mhalf * 32 + l16;
    const int   m1 = m0 + 16;
    const float wq0 = wn2[m0];
    const float wq1 = wn2[m1];
#pragma unroll
    for (int r = 0; r < 8; ++r) {
      // score = ||ws||^2 - 2*WSCALE*(z . ws)  (argmin-equivalent to ref)
      float sc0 = wq0 - 2.0f * WSCALE * acc0[r];
      if (sc0 < minv[r]) { minv[r] = sc0; mini[r] = m0; }
      float sc1 = wq1 - 2.0f * WSCALE * acc1[r];
      if (sc1 < minv[r]) { minv[r] = sc1; mini[r] = m1; }
    }
  }

  // Reduce across the 16 lanes that hold different columns of the same rows.
#pragma unroll
  for (int off = 8; off >= 1; off >>= 1) {
#pragma unroll
    for (int r = 0; r < 8; ++r) {
      float ov = __shfl_xor(minv[r], off, 32);
      int   oi = __shfl_xor(mini[r], off, 32);
      if (ov < minv[r] || (ov == minv[r] && oi < mini[r])) {
        minv[r] = ov; mini[r] = oi;
      }
    }
  }
  if (l16 == 0) {
    const int rb = (lane >> 4) * 8;              // lanes 0/16 -> rows 0-7 / 8-15
#pragma unroll
    for (int r = 0; r < 8; ++r) {
      redv[strip][rb + r][mhalf] = minv[r];
      redi[strip][rb + r][mhalf] = mini[r];
    }
  }
  __syncthreads();
  if (tid < TILE_N) {
    const int s = tid >> 4, row = tid & 15;
    float v0 = redv[s][row][0], v1 = redv[s][row][1];
    int   i0 = redi[s][row][0], i1 = redi[s][row][1];
    idxout[nbase + s * 16 + row] = (v1 < v0 || (v1 == v0 && i1 < i0)) ? i1 : i0;
  }
}

// ---------------------------------------------------------------------------
// Gather z_q = W[idx] into NCHW output (fp32), accumulate sum of squared
// differences, codebook-usage histogram, and index output.
// ---------------------------------------------------------------------------
__global__ void vq_gather(const float* __restrict__ z, const float* __restrict__ W,
                          const int* __restrict__ idxbuf, float* __restrict__ out,
                          float* __restrict__ outIdxF, int* __restrict__ hist,
                          float* __restrict__ sumsq) {
  __shared__ float ls[8];
  const int n = blockIdx.x;
  const int c = threadIdx.x;
  const int k = idxbuf[n];
  const int b = n >> 10, rem = n & 1023;         // rem = h*32 + w
  const size_t zoff = (size_t)(b * EDIM + c) * 1024 + rem;
  const float wq = W[(size_t)k * EDIM + c];
  const float zv = z[zoff];
  out[zoff] = wq;                                // straight-through == z_q
  float d = wq - zv;
  float s = d * d;
#pragma unroll
  for (int off = 16; off >= 1; off >>= 1) s += __shfl_xor(s, off, 32);
  if ((threadIdx.x & 31) == 0) ls[threadIdx.x >> 5] = s;
  __syncthreads();
  if (threadIdx.x == 0) {
    float t = 0.0f;
#pragma unroll
    for (int i = 0; i < 8; ++i) t += ls[i];
    atomicAdd(sumsq, t);
    atomicAdd(&hist[k], 1);
    outIdxF[n] = (float)k;
  }
}

// ---------------------------------------------------------------------------
// Finalize: loss scalar and usage perplexity. Single block.
// ---------------------------------------------------------------------------
__global__ void vq_final(const int* __restrict__ hist, const float* __restrict__ sumsq,
                         float* __restrict__ lossOut, float* __restrict__ perpOut) {
  __shared__ float ls[8];
  float s = 0.0f;
  for (int i = threadIdx.x; i < N_E; i += 256) {
    float p = (float)hist[i] * (1.0f / (float)NTOT);
    s += p * logf(p + 1e-10f);
  }
#pragma unroll
  for (int off = 16; off >= 1; off >>= 1) s += __shfl_xor(s, off, 32);
  if ((threadIdx.x & 31) == 0) ls[threadIdx.x >> 5] = s;
  __syncthreads();
  if (threadIdx.x == 0) {
    float t = 0.0f;
#pragma unroll
    for (int i = 0; i < 8; ++i) t += ls[i];
    *perpOut = expf(-t);
    // loss = mean((zq-z)^2) + beta*mean((zq-z)^2) = 1.25 * SSE / (N*C)
    *lossOut = 1.25f * (*sumsq) * (1.0f / 8388608.0f);
  }
}

// ---------------------------------------------------------------------------
extern "C" void kernel_launch(void* const* d_in, const int* in_sizes, int n_in,
                              void* d_out, int out_size, void* d_ws, size_t ws_size,
                              hipStream_t stream) {
  const float* z = (const float*)d_in[0];        // (32,256,32,32) fp32
  const float* W = (const float*)d_in[1];        // (8192,256) fp32
  float* out = (float*)d_out;                    // [z_q | loss | index | perplexity]

  char* ws = (char*)d_ws;
  _Float16* zf16   = (_Float16*)(ws);                      // 16,777,216 B
  _Float16* wsf16  = (_Float16*)(ws + 16777216);           //  4,194,304 B
  float*    wn2    = (float*)   (ws + 20971520);           //     32,768 B
  int*      idxbuf = (int*)     (ws + 21004288);           //    131,072 B
  int*      hist   = (int*)     (ws + 21135360);           //     32,768 B
  float*    sumsq  = (float*)   (ws + 21168128);           //         16 B

  // zero histogram + loss accumulator (graph-capture safe)
  hipMemsetAsync(hist, 0, N_E * sizeof(int) + 16, stream);

  vq_prep_w<<<N_E / 8, 256, 0, stream>>>(W, wsf16, wn2);
  vq_prep_z<<<1024, 256, 0, stream>>>(z, zf16);
  vq_argmin_wmma<<<NTOT / TILE_N, 256, 0, stream>>>(zf16, wsf16, wn2, idxbuf);

  float* lossOut = out + 8388608;
  float* idxOut  = out + 8388609;
  float* perpOut = out + 8388609 + NTOT;
  vq_gather<<<NTOT, 256, 0, stream>>>(z, W, idxbuf, out, idxOut, hist, sumsq);
  vq_final<<<1, 256, 0, stream>>>(hist, sumsq, lossOut, perpOut);
}
